// EncoderLayer_52115133170008
// MI455X (gfx1250) — compile-verified, matrix operand order checked by reading, not run on hardware
//
#include <hip/hip_runtime.h>
#include <hip/hip_bf16.h>
#include <math.h>

// ---------------------------------------------------------------------------
// EncoderLayer for MI455X (gfx1250): bf16 WMMA GEMMs + fused epilogues.
// B=8, L=1024, D=512, H=8, E=64, DF=2048, NBLK=3
// ---------------------------------------------------------------------------

#define TB 256   // 8 wave32 waves per block

typedef __bf16 bf16_t;
typedef __bf16 v16bf __attribute__((ext_vector_type(16)));
typedef __bf16 bf16x2 __attribute__((ext_vector_type(2)));
typedef float  v8f   __attribute__((ext_vector_type(8)));
typedef unsigned int u32;
typedef u32 u32x2 __attribute__((ext_vector_type(2)));
typedef u32 u32x4 __attribute__((ext_vector_type(4)));

// Native f32->bf16 convert (backend selects v_cvt_*bf16_f32 on gfx1250).
__device__ __forceinline__ bf16_t f2bf(float f) { return (bf16_t)f; }
// pack two floats -> two bf16 in one u32 via hardware convert
__device__ __forceinline__ u32 pkbf(float a, float b) {
  bf16x2 h;
  h[0] = (bf16_t)a;
  h[1] = (bf16_t)b;
  return __builtin_bit_cast(u32, h);
}
// K-permutation so a lane-half's 16 fragment elements are contiguous in LDS:
// storage order [K0..7, K16..23, K8..15, K24..31]
__device__ __forceinline__ int permk(int k) {
  int g = k >> 3;
  int pg = ((g & 1) << 1) | (g >> 1);
  return (k & 7) + (pg << 3);
}

union FragU { u32x4 q[2]; v16bf v; };
__device__ __forceinline__ v16bf load_frag(const bf16_t* p) {
  FragU f;
  f.q[0] = *(const u32x4*)(p);       // 16B aligned by construction
  f.q[1] = *(const u32x4*)(p + 8);
  return f.v;
}

// ---------------------------------------------------------------------------
// Generalized batched GEMM:
//   C[z][M,N] = act( scale * A[z][M,K] x Bmat[z][K,N] + bias + res[z] )
// Batch z decomposes as (z/bdiv, z%bdiv) with outer/inner strides (multi-head
// slices b*L*D + h*E).  transB: Bmat stored [N,K] (Q*K^T).
// Tiling: 128x64 block tile, TK=32; 8 waves; wave -> 32x32 slab, 4 WMMA/step.
// ---------------------------------------------------------------------------
__global__ __launch_bounds__(TB)
void gemm_wmma_kernel(const float* __restrict__ A, long lda, long sAo, long sAi,
                      const float* __restrict__ W, long ldw, long sWo, long sWi,
                      const float* __restrict__ bias,
                      const float* __restrict__ res, long ldr, long sRo, long sRi,
                      float* __restrict__ C, long ldc, long sCo, long sCi,
                      int K, int bdiv, float scale, int transB, int relu) {
  const int z  = blockIdx.z;
  const int zo = z / bdiv, zi = z % bdiv;
  A += (long)zo * sAo + (long)zi * sAi;
  W += (long)zo * sWo + (long)zi * sWi;
  C += (long)zo * sCo + (long)zi * sCi;
  if (res) res += (long)zo * sRo + (long)zi * sRi;

  const int bm = blockIdx.y * 128;
  const int bn = blockIdx.x * 64;

  // row stride 40 elems (80B): 16B-aligned fragment reads, conflict-free banks
  __shared__ __attribute__((aligned(16))) bf16_t lA[128][40];
  __shared__ __attribute__((aligned(16))) bf16_t lB[64][40];   // [col][perm k]

  const int tid  = threadIdx.x;
  const int wave = tid >> 5;
  const int lane = tid & 31;
  const int wr   = wave >> 1;      // 0..3 -> 32-row slab
  const int wc   = wave & 1;       // 0..1 -> 32-col slab
  const int lr   = lane & 15;
  const int lh   = lane >> 4;

  // ---- hoisted staging addresses (K-invariant parts) ----
  // A tile 128x32: 1024 float4 loads, 4 per thread
  const float4* pa[4];
  u32x2*        da[4];
  #pragma unroll
  for (int it = 0; it < 4; ++it) {
    int i  = tid + it * TB;
    int r  = i >> 3;
    int c4 = (i & 7) << 2;
    pa[it] = (const float4*)(A + (long)(bm + r) * lda + c4);
    da[it] = (u32x2*)&lA[r][permk(c4)];
  }
  // B tile 32x64 (or 64x32 transposed source): 512 float4 loads, 2 per thread
  const float4* pbt[2];      // transB path (contiguous K in global)
  u32x2*        dbt[2];
  const float*  pbn[2];      // non-trans path (contiguous N in global)
  int           krn[2], c4n[2];
  #pragma unroll
  for (int it = 0; it < 2; ++it) {
    int i = tid + it * TB;
    {  // transB precompute
      int n  = i >> 3;
      int k4 = (i & 7) << 2;
      pbt[it] = (const float4*)(W + (long)(bn + n) * ldw + k4);
      dbt[it] = (u32x2*)&lB[n][permk(k4)];
    }
    {  // non-trans precompute
      int kr = i >> 4;
      int c4 = (i & 15) << 2;
      krn[it] = kr; c4n[it] = c4;
      pbn[it] = W + (long)kr * ldw + bn + c4;
    }
  }

  const int rowA0 = 32 * wr + lr, rowA1 = rowA0 + 16;
  const int colB0 = 32 * wc + lr, colB1 = colB0 + 16;

  v8f c00 = {}, c01 = {}, c10 = {}, c11 = {};

  for (int k0 = 0; k0 < K; k0 += 32) {
    // ---- stage A (f32 -> packed bf16, 8B LDS stores) ----
    #pragma unroll
    for (int it = 0; it < 4; ++it) {
      float4 v = pa[it][k0 >> 2];
      u32x2 pk; pk.x = pkbf(v.x, v.y); pk.y = pkbf(v.z, v.w);
      *da[it] = pk;
    }
    // ---- stage B ----
    if (transB) {
      #pragma unroll
      for (int it = 0; it < 2; ++it) {
        float4 v = pbt[it][k0 >> 2];
        u32x2 pk; pk.x = pkbf(v.x, v.y); pk.y = pkbf(v.z, v.w);
        *dbt[it] = pk;
      }
    } else {
      #pragma unroll
      for (int it = 0; it < 2; ++it) {
        float4 v = *(const float4*)(pbn[it] + (long)k0 * ldw);
        int pk_ = permk(krn[it]);
        int c4  = c4n[it];
        lB[c4 + 0][pk_] = f2bf(v.x);
        lB[c4 + 1][pk_] = f2bf(v.y);
        lB[c4 + 2][pk_] = f2bf(v.z);
        lB[c4 + 3][pk_] = f2bf(v.w);
      }
    }
    // prefetch next K tile (global_prefetch_b8)
    if (k0 + 32 < K) {
      __builtin_prefetch((const void*)(pa[0] + ((k0 + 32) >> 2)), 0, 1);
      if (transB) __builtin_prefetch((const void*)(pbt[0] + ((k0 + 32) >> 2)), 0, 1);
      else        __builtin_prefetch((const void*)(pbn[0] + (long)(k0 + 32) * ldw), 0, 1);
    }
    __syncthreads();

    // ---- fragments: contiguous 32B per lane -> 2x ds_load_b128 each ----
    v16bf a0 = load_frag(&lA[rowA0][lh * 16]);
    v16bf a1 = load_frag(&lA[rowA1][lh * 16]);
    v16bf b0 = load_frag(&lB[colB0][lh * 16]);
    v16bf b1 = load_frag(&lB[colB1][lh * 16]);

    c00 = __builtin_amdgcn_wmma_f32_16x16x32_bf16(false, a0, false, b0, (short)0, c00, false, false);
    c01 = __builtin_amdgcn_wmma_f32_16x16x32_bf16(false, a0, false, b1, (short)0, c01, false, false);
    c10 = __builtin_amdgcn_wmma_f32_16x16x32_bf16(false, a1, false, b0, (short)0, c10, false, false);
    c11 = __builtin_amdgcn_wmma_f32_16x16x32_bf16(false, a1, false, b1, (short)0, c11, false, false);
    __syncthreads();
  }

  // ---- epilogue: D layout row m = base + r + 8*lh, col n = base + lane%16 ----
  const int m0b = bm + 32 * wr;
  const int n0  = bn + 32 * wc + lr;
  const int n1  = n0 + 16;
  #pragma unroll
  for (int r = 0; r < 8; ++r) {
    int m0 = m0b + r + 8 * lh;
    int m1 = m0 + 16;
    float v00 = c00[r] * scale, v01 = c01[r] * scale;
    float v10 = c10[r] * scale, v11 = c11[r] * scale;
    if (bias) { float b0v = bias[n0], b1v = bias[n1];
                v00 += b0v; v10 += b0v; v01 += b1v; v11 += b1v; }
    if (res)  { v00 += res[(long)m0 * ldr + n0]; v01 += res[(long)m0 * ldr + n1];
                v10 += res[(long)m1 * ldr + n0]; v11 += res[(long)m1 * ldr + n1]; }
    if (relu) { v00 = fmaxf(v00, 0.f); v01 = fmaxf(v01, 0.f);
                v10 = fmaxf(v10, 0.f); v11 = fmaxf(v11, 0.f); }
    C[(long)m0 * ldc + n0] = v00;
    C[(long)m0 * ldc + n1] = v01;
    C[(long)m1 * ldc + n0] = v10;
    C[(long)m1 * ldc + n1] = v11;
  }
}

// ---------------------------------------------------------------------------
// Row softmax, in place. One block per row, values held in registers.
// ---------------------------------------------------------------------------
__global__ __launch_bounds__(TB)
void softmax_kernel(float* __restrict__ S, int N) {
  long row = blockIdx.x;
  float* p = S + row * (long)N;
  float vals[8];
  int n = 0;
  float mx = -3.402823466e38f;
  for (int i = threadIdx.x; i < N; i += TB) { float v = p[i]; vals[n++] = v; mx = fmaxf(mx, v); }
  for (int o = 16; o; o >>= 1) mx = fmaxf(mx, __shfl_xor(mx, o, 32));
  __shared__ float rr[TB / 32];
  __shared__ float bc;
  int lane = threadIdx.x & 31, wv = threadIdx.x >> 5;
  if (!lane) rr[wv] = mx;
  __syncthreads();
  if (threadIdx.x == 0) { float m = rr[0]; for (int i = 1; i < TB / 32; ++i) m = fmaxf(m, rr[i]); bc = m; }
  __syncthreads();
  mx = bc;
  float s = 0.f;
  for (int j = 0; j < n; ++j) { float e = __expf(vals[j] - mx); vals[j] = e; s += e; }
  for (int o = 16; o; o >>= 1) s += __shfl_xor(s, o, 32);
  __syncthreads();
  if (!lane) rr[wv] = s;
  __syncthreads();
  if (threadIdx.x == 0) { float t = 0.f; for (int i = 0; i < TB / 32; ++i) t += rr[i]; bc = 1.0f / t; }
  __syncthreads();
  float inv = bc;
  n = 0;
  for (int i = threadIdx.x; i < N; i += TB) p[i] = vals[n++] * inv;
}

// ---------------------------------------------------------------------------
// out[row] = LN(a[row] + (b?b[row]:0)) * g + beta        (one block per row)
// ---------------------------------------------------------------------------
__global__ __launch_bounds__(TB)
void add_ln_kernel(const float* __restrict__ a, const float* __restrict__ b,
                   const float* __restrict__ g, const float* __restrict__ be,
                   float* __restrict__ out, int D) {
  long row = blockIdx.x;
  const float* pa = a + row * (long)D;
  const float* pb = b ? b + row * (long)D : nullptr;
  float vals[8];
  int n = 0;
  float s = 0.f, s2 = 0.f;
  for (int i = threadIdx.x; i < D; i += TB) {
    float v = pa[i] + (pb ? pb[i] : 0.f);
    vals[n++] = v; s += v; s2 += v * v;
  }
  for (int o = 16; o; o >>= 1) { s += __shfl_xor(s, o, 32); s2 += __shfl_xor(s2, o, 32); }
  __shared__ float rs[TB / 32], rs2[TB / 32];
  __shared__ float mb[2];
  int lane = threadIdx.x & 31, wv = threadIdx.x >> 5;
  if (!lane) { rs[wv] = s; rs2[wv] = s2; }
  __syncthreads();
  if (threadIdx.x == 0) {
    float S = 0.f, S2 = 0.f;
    for (int i = 0; i < TB / 32; ++i) { S += rs[i]; S2 += rs2[i]; }
    float mean = S / (float)D;
    float var  = S2 / (float)D - mean * mean;
    mb[0] = mean; mb[1] = rsqrtf(var + 1e-5f);
  }
  __syncthreads();
  float mean = mb[0], inv = mb[1];
  n = 0;
  for (int i = threadIdx.x; i < D; i += TB)
    out[row * (long)D + i] = (vals[n++] - mean) * inv * g[i] + be[i];
}

// ---------------------------------------------------------------------------
// Faithful "transpose(1,2).contiguous().view(-1,L,D)" memory reinterpretation:
// dst[b, i] = src[b, i % L, i / L]   (i in [0, L*D))
// ---------------------------------------------------------------------------
__global__ __launch_bounds__(TB)
void remap_kernel(const float* __restrict__ src, float* __restrict__ dst,
                  int Bn, int Ln, int Dn) {
  long per = (long)Ln * Dn;
  long total = (long)Bn * per;
  for (long idx = (long)blockIdx.x * TB + threadIdx.x; idx < total;
       idx += (long)gridDim.x * TB) {
    long b = idx / per;
    long i = idx - b * per;
    long l = i % Ln;
    long d = i / Ln;
    dst[idx] = src[b * per + l * (long)Dn + d];
  }
}

// ---------------------------------------------------------------------------
extern "C" void kernel_launch(void* const* d_in, const int* in_sizes, int n_in,
                              void* d_out, int out_size, void* d_ws, size_t ws_size,
                              hipStream_t stream) {
  (void)in_sizes; (void)n_in; (void)out_size; (void)ws_size;
  const int  Bn = 8, Ln = 1024, Dn = 512, Hn = 8, En = 64, DFn = 2048, NB = 3;
  const long BLD = (long)Bn * Ln * Dn;

  const float* x    = (const float*)d_in[0];
  const float* aw_q = (const float*)d_in[1];  const float* ab_q = (const float*)d_in[2];
  const float* aw_k = (const float*)d_in[3];  const float* ab_k = (const float*)d_in[4];
  const float* aw_v = (const float*)d_in[5];  const float* ab_v = (const float*)d_in[6];
  const float* aw_o = (const float*)d_in[7];  const float* ab_o = (const float*)d_in[8];
  const float* n1g  = (const float*)d_in[9];  const float* n1b  = (const float*)d_in[10];
  const float* n2g  = (const float*)d_in[11]; const float* n2b  = (const float*)d_in[12];
  const float* tqw  = (const float*)d_in[13]; const float* tqb  = (const float*)d_in[14];
  const float* tkw  = (const float*)d_in[15]; const float* tkb  = (const float*)d_in[16];
  const float* tvw  = (const float*)d_in[17]; const float* tvb  = (const float*)d_in[18];
  const float* tow  = (const float*)d_in[19]; const float* tob  = (const float*)d_in[20];
  const float* tf1w = (const float*)d_in[21]; const float* tf1b = (const float*)d_in[22];
  const float* tf2w = (const float*)d_in[23]; const float* tf2b = (const float*)d_in[24];
  const float* tng  = (const float*)d_in[25]; const float* tnb  = (const float*)d_in[26];

  float* out0 = (float*)d_out;          // [B,L,D]
  float* Aout = out0 + BLD;             // [B,H,L,L] attention probs (output #2)

  float* w    = (float*)d_ws;
  float* qb   = w;                      // [B,L,D]
  float* kb   = w + BLD;                // [B,L,D]
  float* vb   = w + 2 * BLD;            // [B,L,D]
  float* sb   = w + 3 * BLD;            // [B,L,L] = 2*BLD
  float* ffb  = w;                      // [B,L,DF] = 4*BLD, aliases q/k/v/s (dead then)
  float* abuf = w + 5 * BLD;
  float* bbuf = w + 6 * BLD;
  float* ybuf = w + 7 * BLD;
  float* x1   = w + 8 * BLD;
  float* xnb  = w + 9 * BLD;

  auto gemm = [&](const float* A, long lda, long sAo, long sAi,
                  const float* Wm, long ldw, long sWo, long sWi, int transB,
                  const float* bias,
                  const float* res, long ldr, long sRo, long sRi,
                  float* C, long ldc, long sCo, long sCi,
                  int M, int N, int K, int nb, int bdiv, float scale, int relu) {
    dim3 g((unsigned)(N / 64), (unsigned)(M / 128), (unsigned)nb);
    gemm_wmma_kernel<<<g, TB, 0, stream>>>(A, lda, sAo, sAi, Wm, ldw, sWo, sWi,
                                           bias, res, ldr, sRo, sRi,
                                           C, ldc, sCo, sCi, K, bdiv, scale, transB, relu);
  };

  const long LD = (long)Ln * Dn, LL = (long)Ln * Ln;

  // ---- outer multi-head attention ----
  gemm(x, Dn, 0, 0, aw_q, Dn, 0, 0, 0, ab_q, nullptr, 0, 0, 0, qb, Dn, 0, 0, Bn * Ln, Dn, Dn, 1, 1, 1.f, 0);
  gemm(x, Dn, 0, 0, aw_k, Dn, 0, 0, 0, ab_k, nullptr, 0, 0, 0, kb, Dn, 0, 0, Bn * Ln, Dn, Dn, 1, 1, 1.f, 0);
  gemm(x, Dn, 0, 0, aw_v, Dn, 0, 0, 0, ab_v, nullptr, 0, 0, 0, vb, Dn, 0, 0, Bn * Ln, Dn, Dn, 1, 1, 1.f, 0);
  // scores (per b,h): Aout = softmax(Q K^T / 8)    -> written straight into d_out
  gemm(qb, Dn, LD, En, kb, Dn, LD, En, 1, nullptr, nullptr, 0, 0, 0,
       Aout, Ln, (long)Hn * LL, LL, Ln, Ln, En, Bn * Hn, Hn, 0.125f, 0);
  softmax_kernel<<<Bn * Hn * Ln, TB, 0, stream>>>(Aout, Ln);
  // ctx = A @ V (heads interleave back into [B,L,D])
  gemm(Aout, Ln, (long)Hn * LL, LL, vb, Dn, LD, En, 0, nullptr, nullptr, 0, 0, 0,
       abuf, Dn, LD, En, Ln, En, Ln, Bn * Hn, Hn, 1.f, 0);
  gemm(abuf, Dn, 0, 0, aw_o, Dn, 0, 0, 0, ab_o, nullptr, 0, 0, 0, bbuf, Dn, 0, 0,
       Bn * Ln, Dn, Dn, 1, 1, 1.f, 0);
  add_ln_kernel<<<Bn * Ln, TB, 0, stream>>>(x, bbuf, n1g, n1b, x1, Dn);

  // ---- 3 temporal attention blocks ----
  const float* ycur = x1;
  for (int i = 0; i < NB; ++i) {
    const float* wq = tqw + (long)i * Dn * Dn; const float* bq = tqb + (long)i * Dn;
    const float* wk = tkw + (long)i * Dn * Dn; const float* bk = tkb + (long)i * Dn;
    const float* wv = tvw + (long)i * Dn * Dn; const float* bv = tvb + (long)i * Dn;
    const float* wo = tow + (long)i * Dn * Dn; const float* bo = tob + (long)i * Dn;

    gemm(ycur, Dn, 0, 0, wq, Dn, 0, 0, 0, bq, nullptr, 0, 0, 0, qb, Dn, 0, 0, Bn * Ln, Dn, Dn, 1, 1, 1.f, 0);
    gemm(ycur, Dn, 0, 0, wk, Dn, 0, 0, 0, bk, nullptr, 0, 0, 0, kb, Dn, 0, 0, Bn * Ln, Dn, Dn, 1, 1, 1.f, 0);
    gemm(ycur, Dn, 0, 0, wv, Dn, 0, 0, 0, bv, nullptr, 0, 0, 0, vb, Dn, 0, 0, Bn * Ln, Dn, Dn, 1, 1, 1.f, 0);

    // S = Q K^T (no scaling, as in the original); softmax; out = S V
    gemm(qb, Dn, LD, 0, kb, Dn, LD, 0, 1, nullptr, nullptr, 0, 0, 0,
         sb, Ln, LL, 0, Ln, Ln, Dn, Bn, 1, 1.f, 0);
    softmax_kernel<<<Bn * Ln, TB, 0, stream>>>(sb, Ln);
    gemm(sb, Ln, LL, 0, vb, Dn, LD, 0, 0, nullptr, nullptr, 0, 0, 0,
         abuf, Dn, LD, 0, Ln, Dn, Ln, Bn, 1, 1.f, 0);

    // faithful bad reshape [B,D,L] -> [B,L,D]
    remap_kernel<<<4096, TB, 0, stream>>>(abuf, bbuf, Bn, Ln, Dn);

    // y = ycur + remapped @ Wo + bo
    gemm(bbuf, Dn, 0, 0, wo, Dn, 0, 0, 0, bo, ycur, Dn, 0, 0, ybuf, Dn, 0, 0,
         Bn * Ln, Dn, Dn, 1, 1, 1.f, 0);
    ycur = ybuf;

    // FFN: y += relu(LN(y) @ W1 + b1) @ W2 + b2
    add_ln_kernel<<<Bn * Ln, TB, 0, stream>>>(ybuf, nullptr, tng + (long)i * Dn, tnb + (long)i * Dn, xnb, Dn);
    gemm(xnb, Dn, 0, 0, tf1w + (long)i * Dn * DFn, DFn, 0, 0, 0, tf1b + (long)i * DFn,
         nullptr, 0, 0, 0, ffb, DFn, 0, 0, Bn * Ln, DFn, Dn, 1, 1, 1.f, 1);
    gemm(ffb, DFn, 0, 0, tf2w + (long)i * DFn * Dn, Dn, 0, 0, 0, tf2b + (long)i * Dn,
         ybuf, Dn, 0, 0, ybuf, Dn, 0, 0, Bn * Ln, Dn, DFn, 1, 1, 1.f, 0);
  }

  // final LN(x1 + y)
  add_ln_kernel<<<Bn * Ln, TB, 0, stream>>>(x1, ybuf, n2g, n2b, out0, Dn);
}
